// DaleConstrainedIntegrator_53051436040340
// MI455X (gfx1250) — matrix operand order repaired and checked by reading, not compiled, not run on hardware
//
#include <hip/hip_runtime.h>

// Problem constants (from reference)
#define N_UNITS 2048
#define BATCH   128
#define TSTEPS  1024
#define NWG     32      // persistent workgroups (grid-sync'ed per step)
#define WGSIZE  256     // 8 wave32s

typedef __bf16 v16bf __attribute__((ext_vector_type(16)));
typedef float  v8f   __attribute__((ext_vector_type(8)));
typedef unsigned short ushort8v __attribute__((ext_vector_type(8)));

__device__ __forceinline__ unsigned short f2bf(float f) {
  unsigned u = __builtin_bit_cast(unsigned, f);
  u += 0x7FFFu + ((u >> 16) & 1u);          // round-to-nearest-even
  return (unsigned short)(u >> 16);
}

// 16-bit fragment: elements 0..7 and 8..15 are two contiguous 8-element runs
union Frag16 { v16bf v; ushort8v h[2]; };

// ---------------------------------------------------------------------------
// Init: zero outputs (we accumulate atomically), broadcast state0 into the
// bf16 state buffer, reset the grid-barrier counter.  B*N == 2*B*T == 262144.
// ---------------------------------------------------------------------------
__global__ void dale_init_kernel(float* __restrict__ out,
                                 unsigned short* __restrict__ stateA,
                                 const float* __restrict__ state0,
                                 unsigned* __restrict__ bar) {
  int i = blockIdx.x * blockDim.x + threadIdx.x;
  if (i < 2 * BATCH * TSTEPS) out[i] = 0.0f;
  if (i < BATCH * N_UNITS)    stateA[i] = f2bf(state0[i & (N_UNITS - 1)]);
  if (i == 0)                 *bar = 0u;
}

// ---------------------------------------------------------------------------
// Prep 1: Weff with Dale signs folded in, cast to bf16.
//   Wb[n][k] = bf16( W[n][k] * signs[k] )   (row-major, k contiguous)
// new_state[b,n] = sum_k A[b,k] * Wb[n][k]  -> B-fragment rows are contiguous.
// ---------------------------------------------------------------------------
__global__ void dale_prep_w_kernel(const float* __restrict__ W,
                                   const float* __restrict__ signs,
                                   unsigned short* __restrict__ Wb) {
  int i = blockIdx.x * blockDim.x + threadIdx.x;   // i in [0, N*N)
  int k = i & (N_UNITS - 1);
  Wb[i] = f2bf(W[i] * signs[k]);
}

// ---------------------------------------------------------------------------
// Prep 2: fold the rank-2 input term into per-column f32 vectors:
//   c0[n] = sum_k mask[k]*enc0[k]*signs[k]*W[n,k]
//   c1[n] = sum_k mask[k]*enc1[k]*signs[k]*W[n,k]
// Then per step:  pre[b,n] = (state @ WbT)[b,n] + x0[b]*c0[n] + x1[b]*c1[n]
// This removes ALL ext/bf16-convert VALU work from the hot K-loop.
// ---------------------------------------------------------------------------
__global__ __launch_bounds__(256)
void dale_prep_c_kernel(const float* __restrict__ W,
                        const float* __restrict__ signs,
                        const float* __restrict__ maskv,
                        const float* __restrict__ enc0,
                        const float* __restrict__ enc1,
                        float* __restrict__ c0, float* __restrict__ c1) {
  __shared__ float r0[8], r1[8];
  const int n   = blockIdx.x;
  const int tid = threadIdx.x;
  const float* __restrict__ row = W + (size_t)n * N_UNITS;
  float p0 = 0.0f, p1 = 0.0f;
  for (int k = tid; k < N_UNITS; k += 256) {
    float wk = row[k] * signs[k] * maskv[k];
    p0 += wk * enc0[k];
    p1 += wk * enc1[k];
  }
#pragma unroll
  for (int off = 1; off < 32; off <<= 1) {
    p0 += __shfl_xor(p0, off, 32);
    p1 += __shfl_xor(p1, off, 32);
  }
  if ((tid & 31) == 0) { r0[tid >> 5] = p0; r1[tid >> 5] = p1; }
  __syncthreads();
  if (tid == 0) {
    float a = 0.0f, b = 0.0f;
#pragma unroll
    for (int i = 0; i < 8; ++i) { a += r0[i]; b += r1[i]; }
    c0[n] = a; c1[n] = b;
  }
}

// ---------------------------------------------------------------------------
// Persistent recurrent integrator.
//   Grid: 32 WGs x 256 threads. WG g owns output columns [64g, 64g+64).
//   Wave w owns batch rows [16w, 16w+16) -> 4 wmma accumulators (16x16 each).
//   Hot K-loop: pure global_load_b128 -> v_wmma_f32_16x16x32_bf16 (no VALU),
//   with WGP-scope prefetch of the next W chunk into WGP$ (shared by all 8
//   waves of the WGP, which read identical B fragments).
//   Epilogue: + x0*c0[n] + x1*c1[n], clip, mask, bf16 state store, decoder
//   dot-products via shfl_xor butterfly + atomic f32 adds.
//   Steps separated by a device-wide barrier (monotonic atomic counter).
// ---------------------------------------------------------------------------
__global__ __launch_bounds__(WGSIZE)
void dale_integrate_kernel(const float* __restrict__ x0,
                           const float* __restrict__ x1,
                           const float* __restrict__ dec0,
                           const float* __restrict__ dec1,
                           const float* __restrict__ maskv,
                           const float* __restrict__ c0g,
                           const float* __restrict__ c1g,
                           const unsigned short* __restrict__ Wb,
                           unsigned short* __restrict__ stateA,
                           unsigned short* __restrict__ stateB,
                           unsigned* __restrict__ bar,
                           float* __restrict__ out) {
  __shared__ float s_c0[N_UNITS], s_c1[N_UNITS];
  __shared__ float s_dec0[N_UNITS], s_dec1[N_UNITS], s_mask[N_UNITS];
  __shared__ float s_x0[BATCH], s_x1[BATCH];

  const int tid = threadIdx.x;
  for (int i = tid; i < N_UNITS; i += WGSIZE) {
    s_c0[i]   = c0g[i];
    s_c1[i]   = c1g[i];
    s_dec0[i] = dec0[i];
    s_dec1[i] = dec1[i];
    s_mask[i] = maskv[i];
  }

  const int w    = tid >> 5;        // wave id in WG (0..7)
  const int L    = tid & 31;        // lane id
  const int ln   = L & 15;
  const int half = L >> 4;          // 0: lanes 0-15, 1: lanes 16-31
  const int n0   = blockIdx.x * 64; // column slab base
  const int mA   = (w << 4) + ln;   // batch row this lane feeds into A-frag
  const int kOffA = half << 4;      // contiguous A k order used here: lanes 16-31 hold K+16..31
  const int kOffB = half << 4;      // contiguous B k order used here: lanes 16-31 hold K+16..31

  float* __restrict__ out0 = out;
  float* __restrict__ out1 = out + BATCH * TSTEPS;

  for (int t = 0; t < TSTEPS; ++t) {
    const unsigned short* __restrict__ sIn  = (t & 1) ? stateB : stateA;
    unsigned short*       __restrict__ sOut = (t & 1) ? stateA : stateB;

    // Stage this step's scalar inputs into LDS (broadcast to epilogue).
    if (tid < BATCH) {
      s_x0[tid] = x0[tid * TSTEPS + t];
      s_x1[tid] = x1[tid * TSTEPS + t];
    }
    __syncthreads();

    v8f acc0 = {}, acc1 = {}, acc2 = {}, acc3 = {};

    const unsigned short* __restrict__ aRow  = sIn + (unsigned)mA * N_UNITS + kOffA;
    const unsigned short* __restrict__ bRow  = Wb + (unsigned)(n0 + ln) * N_UNITS + kOffB;

    for (int k0 = 0; k0 < N_UNITS; k0 += 32) {
      // A fragment: raw bf16 state, contiguous k order k = 16*half + s, same as B (no VALU).
      Frag16 ap;
      ap.h[0] = *(const ushort8v*)(aRow + k0);        // K = kb .. kb+7
      ap.h[1] = *(const ushort8v*)(aRow + k0 + 8);    // K = kb+8 .. kb+15

      // B fragments: 4 tiles (n0 + 16c), contiguous 32B per lane.
      const unsigned short* bBase = bRow + k0;
      // WGP-scope prefetch: pull next chunk of W into WGP$, which all 8
      // waves of this WGP hit (they read identical B fragments).
      __builtin_prefetch(bBase + 32, 0, 3);

      Frag16 b0, b1, b2, b3;
      b0.h[0] = *(const ushort8v*)(bBase);
      b0.h[1] = *(const ushort8v*)(bBase + 8);
      b1.h[0] = *(const ushort8v*)(bBase + 16 * N_UNITS);
      b1.h[1] = *(const ushort8v*)(bBase + 16 * N_UNITS + 8);
      b2.h[0] = *(const ushort8v*)(bBase + 32 * N_UNITS);
      b2.h[1] = *(const ushort8v*)(bBase + 32 * N_UNITS + 8);
      b3.h[0] = *(const ushort8v*)(bBase + 48 * N_UNITS);
      b3.h[1] = *(const ushort8v*)(bBase + 48 * N_UNITS + 8);

      acc0 = __builtin_amdgcn_wmma_f32_16x16x32_bf16(false, ap.v, false, b0.v,
                                                     (short)0, acc0, false, false);
      acc1 = __builtin_amdgcn_wmma_f32_16x16x32_bf16(false, ap.v, false, b1.v,
                                                     (short)0, acc1, false, false);
      acc2 = __builtin_amdgcn_wmma_f32_16x16x32_bf16(false, ap.v, false, b2.v,
                                                     (short)0, acc2, false, false);
      acc3 = __builtin_amdgcn_wmma_f32_16x16x32_bf16(false, ap.v, false, b3.v,
                                                     (short)0, acc3, false, false);
    }

    // ---- Epilogue: rank-2 correction, clip, mask, bf16 store, decoders ----
#pragma unroll
    for (int r = 0; r < 8; ++r) {
      const int mOut = (w << 4) + r + (half << 3);  // C/D layout: M=r (+8 hi half)
      const float xs0 = s_x0[mOut];
      const float xs1 = s_x1[mOut];
      const int nc0 = n0 + ln, nc1 = n0 + 16 + ln, nc2 = n0 + 32 + ln, nc3 = n0 + 48 + ln;

      float u0 = acc0[r] + xs0 * s_c0[nc0] + xs1 * s_c1[nc0];
      float u1 = acc1[r] + xs0 * s_c0[nc1] + xs1 * s_c1[nc1];
      float u2 = acc2[r] + xs0 * s_c0[nc2] + xs1 * s_c1[nc2];
      float u3 = acc3[r] + xs0 * s_c0[nc3] + xs1 * s_c1[nc3];

      float v0 = fminf(fmaxf(u0, 0.0f), 1.0e8f) * s_mask[nc0];
      float v1 = fminf(fmaxf(u1, 0.0f), 1.0e8f) * s_mask[nc1];
      float v2 = fminf(fmaxf(u2, 0.0f), 1.0e8f) * s_mask[nc2];
      float v3 = fminf(fmaxf(u3, 0.0f), 1.0e8f) * s_mask[nc3];

      unsigned short* so = sOut + (unsigned)mOut * N_UNITS;
      so[nc0] = f2bf(v0);
      so[nc1] = f2bf(v1);
      so[nc2] = f2bf(v2);
      so[nc3] = f2bf(v3);

      float p0 = s_dec0[nc0] * v0 + s_dec0[nc1] * v1 + s_dec0[nc2] * v2 + s_dec0[nc3] * v3;
      float p1 = s_dec1[nc0] * v0 + s_dec1[nc1] * v1 + s_dec1[nc2] * v2 + s_dec1[nc3] * v3;
#pragma unroll
      for (int off = 1; off < 16; off <<= 1) {   // 16-lane butterfly (stays in half)
        p0 += __shfl_xor(p0, off, 32);
        p1 += __shfl_xor(p1, off, 32);
      }
      if (ln == 0) {
        atomicAdd(&out0[mOut * TSTEPS + t], p0);
        atomicAdd(&out1[mOut * TSTEPS + t], p1);
      }
    }

    // ---- Device-wide step barrier (monotonic counter, no reset races) -----
    __threadfence();
    __syncthreads();
    if (tid == 0) {
      __hip_atomic_fetch_add(bar, 1u, __ATOMIC_ACQ_REL, __HIP_MEMORY_SCOPE_AGENT);
      const unsigned target = (unsigned)NWG * (unsigned)(t + 1);
      while (__hip_atomic_load(bar, __ATOMIC_ACQUIRE, __HIP_MEMORY_SCOPE_AGENT) < target) {
        __builtin_amdgcn_s_sleep(2);
      }
    }
    __syncthreads();
    __threadfence();
  }
}

// ---------------------------------------------------------------------------
// Launch wrapper
//   inputs: x0, x1, enc0, enc1, dec0, dec1, W, signs, mask, state0
//   out: [o0 (B*T), o1 (B*T)] fp32
//   ws: [bar(256B) | stateA(512KB bf16) | stateB(512KB bf16) | Wb(8MB bf16)
//        | c0(8KB f32) | c1(8KB f32)]
// ---------------------------------------------------------------------------
extern "C" void kernel_launch(void* const* d_in, const int* in_sizes, int n_in,
                              void* d_out, int out_size, void* d_ws, size_t ws_size,
                              hipStream_t stream) {
  (void)in_sizes; (void)n_in; (void)out_size; (void)ws_size;

  const float* x0     = (const float*)d_in[0];
  const float* x1     = (const float*)d_in[1];
  const float* enc0   = (const float*)d_in[2];
  const float* enc1   = (const float*)d_in[3];
  const float* dec0   = (const float*)d_in[4];
  const float* dec1   = (const float*)d_in[5];
  const float* W      = (const float*)d_in[6];
  const float* signs  = (const float*)d_in[7];
  const float* maskv  = (const float*)d_in[8];
  const float* state0 = (const float*)d_in[9];
  float* out          = (float*)d_out;

  uintptr_t base = (uintptr_t)d_ws;
  unsigned*       bar    = (unsigned*)base;
  unsigned short* stateA = (unsigned short*)(base + 256);
  unsigned short* stateB = stateA + (size_t)BATCH * N_UNITS;
  unsigned short* Wb     = stateB + (size_t)BATCH * N_UNITS;
  float*          c0     = (float*)(Wb + (size_t)N_UNITS * N_UNITS);
  float*          c1     = c0 + N_UNITS;

  // 262144 threads covers both 2*B*T and B*N
  dale_init_kernel<<<(2 * BATCH * TSTEPS + 255) / 256, 256, 0, stream>>>(
      out, stateA, state0, bar);

  dale_prep_w_kernel<<<(N_UNITS * N_UNITS) / 256, 256, 0, stream>>>(W, signs, Wb);

  dale_prep_c_kernel<<<N_UNITS, 256, 0, stream>>>(W, signs, maskv, enc0, enc1, c0, c1);

  dale_integrate_kernel<<<NWG, WGSIZE, 0, stream>>>(
      x0, x1, dec0, dec1, maskv, c0, c1, Wb, stateA, stateB, bar, out);
}
